// PaiNN_89120571392223
// MI455X (gfx1250) — matrix operand order repaired and checked by reading, not compiled, not run on hardware
//
#include <hip/hip_runtime.h>
#include <hip/hip_bf16.h>

// ---------------------------------------------------------------------------
// PaiNN message layer for MI455X (gfx1250, wave32, WMMA)
//   - bf16 WMMA (f32 accumulate); weights pre-packed into native B layout
//   - LDS A operands stored pre-packed -> ds_load_b128 instead of u16+moves
//   - RBF projection fused into the edge kernel (never materialize [E,384])
//   - async global->LDS staging (ASYNCcnt) + global_prefetch_b8 for gathers
// ---------------------------------------------------------------------------

typedef __attribute__((ext_vector_type(16))) __bf16 v16bf;
typedef __attribute__((ext_vector_type(8)))  float  v8f;

#define HDIM 128
#define RDIM 64

#if __has_builtin(__builtin_amdgcn_global_load_async_to_lds_b128) && \
    __has_builtin(__builtin_amdgcn_s_wait_asynccnt)
#define HAVE_ASYNC_LDS 1
typedef int v4i __attribute__((vector_size(16)));
typedef __attribute__((address_space(1))) v4i GV4;   // global int4*
typedef __attribute__((address_space(3))) v4i LV4;   // LDS int4*
#else
#define HAVE_ASYNC_LDS 0
#endif

__device__ __forceinline__ unsigned short f2bfu(float f) {
  union { float f; unsigned u; } c; c.f = f;
  unsigned r = c.u + 0x7FFFu + ((c.u >> 16) & 1u);   // round-to-nearest-even
  return (unsigned short)(r >> 16);
}

// fast ScaledSiLU: x * sigmoid(x) / 0.6, sigmoid via single v_rcp_f32
__device__ __forceinline__ float scaled_silu(float v) {
  float inv = __builtin_amdgcn_rcpf(1.f + __expf(-v));
  return v * inv * (1.f / 0.6f);
}

// ---- WMMA operand layout helpers (ISA 7.12.2, bf16 16x32 A / 32x16 B) -----
// A element mapping within a K=32 tile: value (row, k) lives at
//   lane = row + 16*k[3],  elem = (k & 7) | ((k & 16) >> 1)
__device__ __forceinline__ int a_lane(int row, int kk) { return row + ((kk & 8) ? 16 : 0); }
__device__ __forceinline__ int a_elem(int kk)          { return (kk & 7) | ((kk & 16) >> 1); }

// Packed tiles: [tile][lane][16 elems] contiguous -> one 32B vector load/lane.
__device__ __forceinline__ v16bf load_A_packed(const unsigned short* ap, int kt, int lane) {
  return *(const v16bf*)(ap + ((((kt << 5) + lane)) << 4));
}
__device__ __forceinline__ v16bf load_B_packed(const unsigned short* wp, int Nt,
                                               int kt, int nt, int lane) {
  return *(const v16bf*)(wp + (((size_t)((kt * Nt + nt) << 5) + lane) << 4));
}

__device__ __forceinline__ v8f wmma_bf16(v16bf a, v16bf b, v8f c) {
  return __builtin_amdgcn_wmma_f32_16x16x32_bf16(
      /*neg_a=*/false, a, /*neg_b=*/false, b,
      /*c_mod=*/(short)0, c, /*reuse_a=*/false, /*reuse_b=*/false);
}

// --------------------------- helper kernels --------------------------------

// Pack row-major f32 W[K][Nn] into bf16 WMMA-B tile layout:
//   dst[((kt*Nt + nt)*32 + lane)*16 + e] = bf16(W[kt*32 + 8*lane[4] + (e&7) + 16*e[3]][nt*16 + lane&15])
__global__ void painn_pack_B(const float* __restrict__ src,
                             unsigned short* __restrict__ dst, int K, int Nn) {
  int idx = blockIdx.x * 256 + threadIdx.x;
  if (idx >= K * Nn) return;
  int e    = idx & 15;
  int lane = (idx >> 4) & 31;
  int tile = idx >> 9;
  int Nt = Nn >> 4;
  int kt = tile / Nt, nt = tile - kt * Nt;
  int k = kt * 32 + ((lane & 16) ? 8 : 0) + (e & 7) + ((e & 8) ? 16 : 0);
  int n = nt * 16 + (lane & 15);
  dst[idx] = f2bfu(src[(size_t)k * Nn + n]);
}

__global__ void painn_zero_f32(float* __restrict__ p, int n4) {
  int i = blockIdx.x * 256 + threadIdx.x;      // one float4 per thread
  if (i < n4) ((float4*)p)[i] = make_float4(0.f, 0.f, 0.f, 0.f);
}

// ----------------------- node kernel: LN + MLP -----------------------------
// One block = 16 nodes, 8 waves. GEMM1 (K=128) -> ScaledSiLU -> GEMM2 -> xh.

__global__ __launch_bounds__(256) void painn_node_kernel(
    const float* __restrict__ x, const float* __restrict__ ln_g,
    const float* __restrict__ ln_b, const unsigned short* __restrict__ W1b,
    const float* __restrict__ b1, const unsigned short* __restrict__ W2b,
    const float* __restrict__ b2, float* __restrict__ xh, int nNodes) {
  __shared__ float          sx[16][HDIM];       // raw tile (8 KB)
  __shared__ unsigned short sxnp[4 * 32 * 16];  // packed A for GEMM1 (4 KB)
  __shared__ unsigned short sh1p[4 * 32 * 16];  // packed A for GEMM2 (4 KB)
  __shared__ float          smu[16], srs[16];

  const int t = threadIdx.x;
  const int rowbase = blockIdx.x * 16;

  // vectorized tile load: 16x128 f32 = 512 float4
  for (int i = t; i < 16 * 32; i += 256) {
    int r = i >> 5, c4 = i & 31;
    int gr = rowbase + r; if (gr >= nNodes) gr = nNodes - 1;
    float4 v = ((const float4*)x)[(size_t)gr * 32 + c4];
    *((float4*)&sx[r][0] + c4) = v;
  }
  __syncthreads();

  if (t < 16) {
    float mu = 0.f;
#pragma unroll 8
    for (int c = 0; c < HDIM; ++c) mu += sx[t][c];
    mu *= (1.f / HDIM);
    float var = 0.f;
#pragma unroll 8
    for (int c = 0; c < HDIM; ++c) { float d = sx[t][c] - mu; var += d * d; }
    smu[t] = mu;
    srs[t] = rsqrtf(var * (1.f / HDIM) + 1e-5f);
  }
  __syncthreads();

  // LayerNorm -> bf16, stored directly in packed-A layout
  for (int i = t; i < 16 * HDIM; i += 256) {
    int r = i >> 7, c = i & 127;
    float xn = (sx[r][c] - smu[r]) * srs[r] * ln_g[c] + ln_b[c];
    int kt = c >> 5, kk = c & 31;
    sxnp[(((kt << 5) + a_lane(r, kk)) << 4) + a_elem(kk)] = f2bfu(xn);
  }
  __syncthreads();

  const int wave = t >> 5, lane = t & 31;
  const int n    = lane & 15;
  const int rofs = (lane & 16) ? 8 : 0;

  // GEMM1: h1[:, 16w:16w+16] = xn @ W1  (K = 128 -> 4 WMMA k-steps)
  v8f acc = {};
#pragma unroll
  for (int ks = 0; ks < 4; ++ks)
    acc = wmma_bf16(load_A_packed(sxnp, ks, lane),
                    load_B_packed(W1b, HDIM / 16, ks, wave, lane), acc);
  {
    const int   col  = wave * 16 + n;
    const float bias = b1[col];
    const int   kt = col >> 5, kk = col & 31;
    const int   e  = a_elem(kk);
#pragma unroll
    for (int i = 0; i < 8; ++i) {
      float s = scaled_silu(acc[i] + bias);
      sh1p[(((kt << 5) + a_lane(i + rofs, kk)) << 4) + e] = f2bfu(s);
    }
  }
  __syncthreads();

  // GEMM2: wave w covers cols {16w, 128+16w, 256+16w} of the 384-wide output
  v16bf a2[4];
#pragma unroll
  for (int ks = 0; ks < 4; ++ks) a2[ks] = load_A_packed(sh1p, ks, lane);

#pragma unroll
  for (int chunk = 0; chunk < 3; ++chunk) {
    const int nt   = chunk * 8 + wave;    // of 24 column tiles
    const int ncol = nt * 16;
    v8f acc2 = {};
#pragma unroll
    for (int ks = 0; ks < 4; ++ks)
      acc2 = wmma_bf16(a2[ks], load_B_packed(W2b, 3 * HDIM / 16, ks, nt, lane), acc2);
    const float bias = b2[ncol + n];
#pragma unroll
    for (int i = 0; i < 8; ++i) {
      int gr = rowbase + i + rofs;
      if (gr < nNodes) xh[(size_t)gr * (3 * HDIM) + ncol + n] = acc2[i] + bias;
    }
  }
}

// ----------------------- edge kernel: fused RBF GEMM + message -------------
// One block = 16 edges. Wave w owns H-column slice [16w,16w+16) of all three
// chunks, so x_m / xh2 / xh3 for a column live in the same lane. rbfh is
// computed in-register (2 WMMAs per chunk, K = 64) and never hits memory.

__global__ __launch_bounds__(256) void painn_edge_kernel(
    const int* __restrict__ edge_index, const float* __restrict__ edge_rbf,
    const float* __restrict__ edge_vector,
    const unsigned short* __restrict__ Wrb, const float* __restrict__ br,
    const float* __restrict__ xh, const float* __restrict__ vecp,
    float* __restrict__ dx, float* __restrict__ dvec, int nEdges) {
  __shared__ float          sraw[16 * RDIM];    // staged raw rbf tile (4 KB)
  __shared__ unsigned short srbfp[2 * 32 * 16]; // packed A (2 KB)
  __shared__ int   ssrc[16], sdst[16];
  __shared__ float sev[16][3];

  const int t  = threadIdx.x;
  const int e0 = blockIdx.x * 16;

  if (t < 16) {
    int e = e0 + t; if (e >= nEdges) e = nEdges - 1;
    ssrc[t] = edge_index[e];            // row 0: src
    sdst[t] = edge_index[nEdges + e];   // row 1: dst
  }
  if (t >= 16 && t < 64) {
    int i = t - 16;                      // 48 values: [16 edges][3]
    int e = e0 + i / 3; if (e >= nEdges) e = nEdges - 1;
    sev[i / 3][i % 3] = edge_vector[(size_t)e * 3 + (i % 3)];
  }

  // Stage the 16x64 f32 rbf tile into LDS (async DMA path on CDNA5).
  {
    int r = t >> 4, q = t & 15;          // row, 16B chunk (4 floats) in row
    int e = e0 + r; if (e >= nEdges) e = nEdges - 1;
    const float* gsrc = edge_rbf + (size_t)e * RDIM + q * 4;
    float*       ldst = sraw + r * RDIM + q * 4;
#if HAVE_ASYNC_LDS
    __builtin_amdgcn_global_load_async_to_lds_b128((GV4*)(void*)gsrc,
                                                   (LV4*)(void*)ldst, 0, 0);
    __builtin_amdgcn_s_wait_asynccnt(0);
#else
    *(float4*)ldst = *(const float4*)gsrc;
#endif
  }
  __syncthreads();

  // convert + pack into WMMA A layout
  for (int i = t; i < 16 * RDIM; i += 256) {
    int r = i >> 6, c = i & 63;
    int kt = c >> 5, kk = c & 31;
    srbfp[(((kt << 5) + a_lane(r, kk)) << 4) + a_elem(kk)] = f2bfu(sraw[i]);
  }

  // Prefetch the gathered xh/vec rows (1536 B each) behind the WMMA work.
  {
    int e = t >> 4, part = t & 15;       // 16 edges x 16 x 96B slices
    const char* px = (const char*)(xh   + (size_t)ssrc[e] * (3 * HDIM)) + part * 96;
    const char* pv = (const char*)(vecp + (size_t)ssrc[e] * (3 * HDIM)) + part * 96;
    __builtin_prefetch(px, 0, 1);
    __builtin_prefetch(pv, 0, 1);
  }
  __syncthreads();

  const int wave = t >> 5, lane = t & 31;
  const int n    = lane & 15;
  const int rofs = (lane & 16) ? 8 : 0;
  const int hcol = wave * 16 + n;

  v16bf a0 = load_A_packed(srbfp, 0, lane);
  v16bf a1 = load_A_packed(srbfp, 1, lane);

  v8f m[3];
#pragma unroll
  for (int chunk = 0; chunk < 3; ++chunk) {
    const int nt   = chunk * 8 + wave;   // of 24 column tiles in [384]
    const int ncol = nt * 16;
    v8f acc = {};
    acc = wmma_bf16(a0, load_B_packed(Wrb, 3 * HDIM / 16, 0, nt, lane), acc);
    acc = wmma_bf16(a1, load_B_packed(Wrb, 3 * HDIM / 16, 1, nt, lane), acc);
    const float bias = br[ncol + n];
#pragma unroll
    for (int i = 0; i < 8; ++i) {
      int row = i + rofs;
      float xhv = xh[(size_t)ssrc[row] * (3 * HDIM) + ncol + n];  // gather
      m[chunk][i] = (acc[i] + bias) * xhv;                        // m = xh[src]*rbfh
    }
  }

  const float inv_sqrt_3 = 0.57735026918962584f;
  const float inv_sqrt_h = 0.08838834764831845f;  // 1/sqrt(128)

#pragma unroll
  for (int i = 0; i < 8; ++i) {
    const int row = i + rofs;
    if (e0 + row >= nEdges) continue;
    const int s = ssrc[row], d = sdst[row];
    atomicAdd(&dx[(size_t)d * HDIM + hcol], m[0][i]);
    const float xh2 = m[1][i] * inv_sqrt_3;
    const float xh3 = m[2][i];
#pragma unroll
    for (int dim = 0; dim < 3; ++dim) {
      float vv  = vecp[(size_t)s * (3 * HDIM) + dim * HDIM + hcol];  // gather
      float out = (vv * xh2 + xh3 * sev[row][dim]) * inv_sqrt_h;
      atomicAdd(&dvec[((size_t)d * 3 + dim) * HDIM + hcol], out);
    }
  }
}

// --------------------------------- launch ----------------------------------

extern "C" void kernel_launch(void* const* d_in, const int* in_sizes, int n_in,
                              void* d_out, int out_size, void* d_ws, size_t ws_size,
                              hipStream_t stream) {
  const float* x    = (const float*)d_in[0];
  const float* vecp = (const float*)d_in[1];
  const int*   ei   = (const int*)d_in[2];
  const float* rbf  = (const float*)d_in[3];
  const float* ev   = (const float*)d_in[4];
  const float* ln_g = (const float*)d_in[5];
  const float* ln_b = (const float*)d_in[6];
  const float* W1   = (const float*)d_in[7];
  const float* b1   = (const float*)d_in[8];
  const float* W2   = (const float*)d_in[9];
  const float* b2   = (const float*)d_in[10];
  const float* Wr   = (const float*)d_in[11];
  const float* br   = (const float*)d_in[12];

  const int N = in_sizes[0] / HDIM;     // 25000
  const int E = in_sizes[3] / RDIM;     // 400000

  float* out  = (float*)d_out;
  float* dx   = out;
  float* dvec = out + (size_t)N * HDIM;

  // workspace: xh f32 [N,384] + packed bf16 weights
  char*  ws  = (char*)d_ws;
  float* xh  = (float*)ws;
  size_t off = (size_t)N * (3 * HDIM) * sizeof(float);
  off = (off + 255) & ~(size_t)255;
  unsigned short* W1b = (unsigned short*)(ws + off); off += (size_t)HDIM * HDIM * 2;
  off = (off + 255) & ~(size_t)255;
  unsigned short* W2b = (unsigned short*)(ws + off); off += (size_t)HDIM * 3 * HDIM * 2;
  off = (off + 255) & ~(size_t)255;
  unsigned short* Wrb = (unsigned short*)(ws + off);
  (void)ws_size; (void)n_in;

  const int nW1 = HDIM * HDIM, nW2 = HDIM * 3 * HDIM, nWr = RDIM * 3 * HDIM;
  painn_pack_B<<<(nW1 + 255) / 256, 256, 0, stream>>>(W1, W1b, HDIM, HDIM);
  painn_pack_B<<<(nW2 + 255) / 256, 256, 0, stream>>>(W2, W2b, HDIM, 3 * HDIM);
  painn_pack_B<<<(nWr + 255) / 256, 256, 0, stream>>>(Wr, Wrb, RDIM, 3 * HDIM);
  // out_size = N*(H + 3H) floats, divisible by 4 -> zero as float4
  const int n4 = out_size >> 2;
  painn_zero_f32<<<(n4 + 255) / 256, 256, 0, stream>>>(out, n4);

  painn_node_kernel<<<(N + 15) / 16, 256, 0, stream>>>(
      x, ln_g, ln_b, W1b, b1, W2b, b2, xh, N);
  painn_edge_kernel<<<(E + 15) / 16, 256, 0, stream>>>(
      ei, rbf, ev, Wrb, br, xh, vecp, dx, dvec, E);
}
